// MCA_20916490732185
// MI455X (gfx1250) — matrix operand order repaired and checked by reading, not compiled, not run on hardware
//
#include <hip/hip_runtime.h>
#include <hip/hip_bf16.h>

typedef __bf16 bf16_t;
typedef __attribute__((ext_vector_type(16))) __bf16 v16bf;
typedef __attribute__((ext_vector_type(8)))  __bf16 v8bf;
typedef __attribute__((ext_vector_type(4)))  __bf16 v4bf;
typedef __attribute__((ext_vector_type(8)))  float  v8f;
typedef int v4i __attribute__((vector_size(16)));   // int4 vector for async-LDS builtin

#define LOG2E 1.4426950408889634f

// ---- gfx1250 async global->LDS path (guarded; falls back to VGPR staging) ----
#if defined(__has_builtin)
#  if __has_builtin(__builtin_amdgcn_global_load_async_to_lds_b128) && \
      __has_builtin(__builtin_amdgcn_s_wait_asynccnt)
#    define HAVE_ASYNC_LDS 1
#  endif
#endif
#ifndef HAVE_ASYNC_LDS
#  define HAVE_ASYNC_LDS 0
#endif

__device__ static inline void copy16_g2s(const void* g, void* l) {
#if HAVE_ASYNC_LDS
  // generic LDS ptr bits[31:0] are the LDS byte address (aperture in [63:32]),
  // so integer truncation to an AS(3) pointer is exact.
  __builtin_amdgcn_global_load_async_to_lds_b128(
      (__attribute__((address_space(1))) v4i*)(unsigned long long)g,
      (__attribute__((address_space(3))) v4i*)(unsigned long long)l,
      0, 0);
#else
  *(uint4*)l = *(const uint4*)g;
#endif
}
__device__ static inline void copy_join() {
#if HAVE_ASYNC_LDS
  __builtin_amdgcn_s_wait_asynccnt(0);
#endif
}

// ---------------------------------------------------------------- convert
__global__ __launch_bounds__(256) void f32_to_bf16_kernel(
    const float* __restrict__ in, bf16_t* __restrict__ out, int n) {
  int i4 = (blockIdx.x * blockDim.x + threadIdx.x) * 4;
  if (i4 < n) {
    float4 v = *(const float4*)(in + i4);
    v4bf o;
    o[0] = (bf16_t)v.x; o[1] = (bf16_t)v.y; o[2] = (bf16_t)v.z; o[3] = (bf16_t)v.w;
    *(v4bf*)(out + i4) = o;
  }
}

// ---------------------------------------------------------------- GEMM
// C[M,N] = A[M,K] @ W[N,K]^T   (A,W bf16; accumulate f32)
// MODE 0: store bf16, head-interleaved out[(col>>6)*M + row][64] with scale
// MODE 1: store f32 row-major + bias
#define BM 128
#define BN 128
#define BK 32
#define LDT (BK + 8)   // 40 elems = 80B rows (16B aligned)

template <int MODE>
__global__ __launch_bounds__(256) void gemm_wmma(
    const bf16_t* __restrict__ A, const bf16_t* __restrict__ W,
    void* __restrict__ Cout, const float* __restrict__ bias,
    float scale, int M, int N, int K) {
  __shared__ __align__(16) bf16_t As[BM][LDT];
  __shared__ __align__(16) bf16_t Bs[BN][LDT];

  const int t    = threadIdx.x;
  const int lane = t & 31;
  const int wid  = t >> 5;     // 8 waves
  const int wm   = wid & 3;    // 4 waves along M (32 rows each)
  const int wn   = wid >> 2;   // 2 waves along N (64 cols each)
  const int half = lane >> 4;  // k-group select (wave32)
  const int l16  = lane & 15;

  const int bm0 = blockIdx.x * BM;
  const int bn0 = blockIdx.y * BN;

  v8f acc[2][4] = {};

  for (int k0 = 0; k0 < K; k0 += BK) {
    // stage A + B tiles: 2x512 b128 chunks, 4 per thread, direct to LDS
#pragma unroll
    for (int i = 0; i < 2; ++i) {
      int c = t + i * 256;
      int r = c >> 2, o = (c & 3) * 8;
      copy16_g2s(A + (size_t)(bm0 + r) * K + k0 + o, &As[r][o]);
      copy16_g2s(W + (size_t)(bn0 + r) * K + k0 + o, &Bs[r][o]);
    }
    copy_join();
    __syncthreads();

    v16bf af[2], bfr[4];
#pragma unroll
    for (int i = 0; i < 2; ++i) {  // A 16x32: elems 0..7 -> k=g*8.., 8..15 -> k=16+g*8..
      const bf16_t* src = &As[wm * 32 + i * 16 + l16][half * 8];
      ((v8bf*)&af[i])[0] = *(const v8bf*)(src);
      ((v8bf*)&af[i])[1] = *(const v8bf*)(src + 16);
    }
#pragma unroll
    for (int j = 0; j < 4; ++j) {  // B 32x16: elems 0..7 -> k=g*8.., 8..15 -> k=16+g*8..
      const bf16_t* src = &Bs[wn * 64 + j * 16 + l16][half * 8];
      ((v8bf*)&bfr[j])[0] = *(const v8bf*)(src);
      ((v8bf*)&bfr[j])[1] = *(const v8bf*)(src + 16);
    }
#pragma unroll
    for (int i = 0; i < 2; ++i)
#pragma unroll
      for (int j = 0; j < 4; ++j)
        acc[i][j] = __builtin_amdgcn_wmma_f32_16x16x32_bf16(
            false, af[i], false, bfr[j], (short)0, acc[i][j], false, false);
    __syncthreads();
  }

  // epilogue: C 16x16 layout -> vgpr r holds row r (+8 for lanes 16-31), col = lane&15
#pragma unroll
  for (int i = 0; i < 2; ++i)
#pragma unroll
    for (int j = 0; j < 4; ++j)
#pragma unroll
      for (int r = 0; r < 8; ++r) {
        int row = bm0 + wm * 32 + i * 16 + half * 8 + r;
        int col = bn0 + wn * 64 + j * 16 + l16;
        float v = acc[i][j][r];
        if (MODE == 0) {
          int h = col >> 6, d = col & 63;
          ((bf16_t*)Cout)[((size_t)h * M + row) * 64 + d] = (bf16_t)(v * scale);
        } else {
          ((float*)Cout)[(size_t)row * N + col] = v + bias[col];
        }
      }
}

// ---------------------------------------------------------------- attention
// Q/K/V: bf16 [HEADS][nTot][64] (Q pre-scaled by SCALE*log2e), out bf16 [nTot][1024]
__global__ __launch_bounds__(128) void attn_fused(
    const bf16_t* __restrict__ Q, const bf16_t* __restrict__ K,
    const bf16_t* __restrict__ V, bf16_t* __restrict__ Oout,
    int seq, int nTot) {
  __shared__ __align__(16) bf16_t Qs[64][72];
  __shared__ __align__(16) bf16_t Ks[64][72];
  __shared__ __align__(16) bf16_t Vt[64][72];       // transposed: [d][m]
  __shared__ __align__(16) bf16_t Ps[4][16][72];    // per-wave P tile

  const int t    = threadIdx.x;
  const int lane = t & 31;
  const int wid  = t >> 5;   // 4 waves x 16 query rows
  const int half = lane >> 4;
  const int l16  = lane & 15;

  const int bh = blockIdx.y;
  const int b  = bh >> 4;
  const int h  = bh & 15;

  const bf16_t* Qp = Q + ((size_t)h * nTot + (size_t)b * seq + blockIdx.x * 64) * 64;
  const bf16_t* Kp = K + ((size_t)h * nTot + (size_t)b * seq) * 64;
  const bf16_t* Vp = V + ((size_t)h * nTot + (size_t)b * seq) * 64;

  // stage Q tile (64x64) straight to LDS
#pragma unroll
  for (int i = 0; i < 4; ++i) {
    int c = t + i * 128;
    int r = c >> 3, o = (c & 7) * 8;
    copy16_g2s(Qp + (size_t)r * 64 + o, &Qs[r][o]);
  }
  copy_join();
  __syncthreads();

  v16bf aq[2];
#pragma unroll
  for (int kk = 0; kk < 2; ++kk) {
    const bf16_t* src = &Qs[wid * 16 + l16][kk * 32 + half * 8];
    ((v8bf*)&aq[kk])[0] = *(const v8bf*)src;
    ((v8bf*)&aq[kk])[1] = *(const v8bf*)(src + 16);
  }

  float mrow[8], lrow[8];
  v8f Oacc[4] = {};
#pragma unroll
  for (int r = 0; r < 8; ++r) { mrow[r] = -INFINITY; lrow[r] = 0.f; }

  for (int m0 = 0; m0 < seq; m0 += 64) {
    __syncthreads();  // previous iteration's Ks/Vt reads done
#pragma unroll
    for (int i = 0; i < 4; ++i) {  // stage K block [m][d] (async direct to LDS)
      int c = t + i * 128;
      int r = c >> 3, o = (c & 7) * 8;
      copy16_g2s(Kp + (size_t)(m0 + r) * 64 + o, &Ks[r][o]);
    }
#pragma unroll
    for (int i = 0; i < 4; ++i) {  // stage V transposed [d][m] (VGPR scatter)
      int c = t + i * 128;
      int r = c >> 3, o = (c & 7) * 8;
      v8bf vv = *(const v8bf*)(Vp + (size_t)(m0 + r) * 64 + o);
#pragma unroll
      for (int e = 0; e < 8; ++e) Vt[o + e][r] = vv[e];
    }
    copy_join();
    __syncthreads();

    // S = Q @ K^T : 4 column tiles of 16, K-dim 64 = 2 wmma steps
    v8f s[4];
#pragma unroll
    for (int j = 0; j < 4; ++j) {
      v8f sj = {};
#pragma unroll
      for (int kk = 0; kk < 2; ++kk) {
        v16bf bk;
        const bf16_t* src = &Ks[j * 16 + l16][kk * 32 + half * 8];
        ((v8bf*)&bk)[0] = *(const v8bf*)src;
        ((v8bf*)&bk)[1] = *(const v8bf*)(src + 16);
        sj = __builtin_amdgcn_wmma_f32_16x16x32_bf16(
            false, aq[kk], false, bk, (short)0, sj, false, false);
      }
      s[j] = sj;
    }

    // online softmax: row slot r lives in lanes of one 16-lane half
#pragma unroll
    for (int r = 0; r < 8; ++r) {
      float v = fmaxf(fmaxf(s[0][r], s[1][r]), fmaxf(s[2][r], s[3][r]));
#pragma unroll
      for (int x = 8; x >= 1; x >>= 1) v = fmaxf(v, __shfl_xor(v, x, 32));
      float mn = fmaxf(mrow[r], v);
      float alpha = exp2f(mrow[r] - mn);
      mrow[r] = mn;
      float rs = 0.f;
#pragma unroll
      for (int j = 0; j < 4; ++j) {
        float p = exp2f(s[j][r] - mn);
        rs += p;
        Ps[wid][half * 8 + r][j * 16 + l16] = (bf16_t)p;
      }
#pragma unroll
      for (int x = 8; x >= 1; x >>= 1) rs += __shfl_xor(rs, x, 32);
      lrow[r] = lrow[r] * alpha + rs;
#pragma unroll
      for (int tt = 0; tt < 4; ++tt) Oacc[tt][r] *= alpha;
    }

    // O += P @ V : P fragments hoisted (depend only on kk), 4 d-tiles
    v16bf ap[2];
#pragma unroll
    for (int kk = 0; kk < 2; ++kk) {
      const bf16_t* ps = &Ps[wid][l16][kk * 32 + half * 8];
      ((v8bf*)&ap[kk])[0] = *(const v8bf*)ps;
      ((v8bf*)&ap[kk])[1] = *(const v8bf*)(ps + 16);
    }
#pragma unroll
    for (int tt = 0; tt < 4; ++tt) {
#pragma unroll
      for (int kk = 0; kk < 2; ++kk) {
        v16bf bv;
        const bf16_t* vs = &Vt[tt * 16 + l16][kk * 32 + half * 8];
        ((v8bf*)&bv)[0] = *(const v8bf*)vs;
        ((v8bf*)&bv)[1] = *(const v8bf*)(vs + 16);
        Oacc[tt] = __builtin_amdgcn_wmma_f32_16x16x32_bf16(
            false, ap[kk], false, bv, (short)0, Oacc[tt], false, false);
      }
    }
  }

  const int rowBase = blockIdx.x * 64 + wid * 16 + half * 8;
#pragma unroll
  for (int tt = 0; tt < 4; ++tt)
#pragma unroll
    for (int r = 0; r < 8; ++r) {
      float v = Oacc[tt][r] / lrow[r];
      int row = rowBase + r;
      int d = tt * 16 + l16;
      Oout[((size_t)b * seq + row) * 1024 + h * 64 + d] = (bf16_t)v;
    }
}

// ---------------------------------------------------------------- launch
extern "C" void kernel_launch(void* const* d_in, const int* in_sizes, int n_in,
                              void* d_out, int out_size, void* d_ws, size_t ws_size,
                              hipStream_t stream) {
  const float* x1 = (const float*)d_in[0];
  const float* x2 = (const float*)d_in[1];
  const float* wq = (const float*)d_in[2];
  const float* wk = (const float*)d_in[3];
  const float* wv = (const float*)d_in[4];
  const float* wo = (const float*)d_in[5];
  const float* bo = (const float*)d_in[6];

  const int B = 2, SEQ = 2048, D = 1024;
  const int M = B * SEQ;  // 4096 rows for all projections

  char* ws = (char*)d_ws;
  size_t off = 0;
  auto alloc = [&](size_t bytes) {
    char* p = ws + off;
    off += (bytes + 255) & ~(size_t)255;
    return p;
  };
  bf16_t* x1b = (bf16_t*)alloc((size_t)M * D * 2);
  bf16_t* x2b = (bf16_t*)alloc((size_t)M * D * 2);
  bf16_t* wqb = (bf16_t*)alloc((size_t)D * D * 2);
  bf16_t* wkb = (bf16_t*)alloc((size_t)D * D * 2);
  bf16_t* wvb = (bf16_t*)alloc((size_t)D * D * 2);
  bf16_t* wob = (bf16_t*)alloc((size_t)D * D * 2);
  bf16_t* Qb  = (bf16_t*)alloc((size_t)M * D * 2);  // [h][4096][64]
  bf16_t* Kb  = (bf16_t*)alloc((size_t)M * D * 2);
  bf16_t* Vb  = (bf16_t*)alloc((size_t)M * D * 2);
  bf16_t* AOb = (bf16_t*)alloc((size_t)M * D * 2);  // [4096][1024]

  const int nMD = M * D, nDD = D * D;
  f32_to_bf16_kernel<<<(nMD / 4 + 255) / 256, 256, 0, stream>>>(x1, x1b, nMD);
  f32_to_bf16_kernel<<<(nMD / 4 + 255) / 256, 256, 0, stream>>>(x2, x2b, nMD);
  f32_to_bf16_kernel<<<(nDD / 4 + 255) / 256, 256, 0, stream>>>(wq, wqb, nDD);
  f32_to_bf16_kernel<<<(nDD / 4 + 255) / 256, 256, 0, stream>>>(wk, wkb, nDD);
  f32_to_bf16_kernel<<<(nDD / 4 + 255) / 256, 256, 0, stream>>>(wv, wvb, nDD);
  f32_to_bf16_kernel<<<(nDD / 4 + 255) / 256, 256, 0, stream>>>(wo, wob, nDD);

  dim3 gg(M / BM, D / BN);  // 32 x 8
  const float qscale = 0.125f * LOG2E;  // HEAD_DIM^-0.5 * log2(e), folded into Q
  gemm_wmma<0><<<gg, 256, 0, stream>>>(x1b, wqb, (void*)Qb, nullptr, qscale, M, D, D);
  gemm_wmma<0><<<gg, 256, 0, stream>>>(x2b, wkb, (void*)Kb, nullptr, 1.0f, M, D, D);
  gemm_wmma<0><<<gg, 256, 0, stream>>>(x2b, wvb, (void*)Vb, nullptr, 1.0f, M, D, D);

  dim3 ga(SEQ / 64, B * 16);  // 32 x 32
  attn_fused<<<ga, 128, 0, stream>>>(Qb, Kb, Vb, AOb, SEQ, M);

  gemm_wmma<1><<<gg, 256, 0, stream>>>(AOb, wob, d_out, bo, 1.0f, M, D, D);
}
